// ChatbotModel_79121887527475
// MI455X (gfx1250) — compile-verified, matrix operand order checked by reading, not run on hardware
//
#include <hip/hip_runtime.h>
#include <cstdint>
#include <cstddef>

// ---------------------------------------------------------------------------
// Types for CDNA5 WMMA (gfx1250, wave32)
// ---------------------------------------------------------------------------
typedef __bf16 bf16;
typedef __attribute__((ext_vector_type(16))) __bf16 v16bf;
typedef __attribute__((ext_vector_type(8)))  float  v8f;

union FragAB { v16bf v; uint4 q[2]; };
union FragC  { v8f   v; float f[8]; };

// Dimensions
#define BB 64
#define SS 20
#define HH 128
#define EE 300
#define VV 50257
#define VROWS_PAD 50272   // 3142 * 16

// ---------------------------------------------------------------------------
// Fragment load: 16-bit A/B 16x32 tile layout (ISA 7.12.2):
//   lanes 0-15  : row r, K = kbase+[0..7]  and kbase+[16..23]
//   lanes 16-31 : row r, K = kbase+[8..15] and kbase+[24..31]
// Source is row-major [16 rows][stride] bf16, 16B-aligned rows.
// ---------------------------------------------------------------------------
__device__ __forceinline__ v16bf load_frag(const bf16* tile, int stride, int lane, int kbase) {
    int r  = lane & 15;
    int hi = lane >> 4;
    const bf16* p = tile + (size_t)r * stride + kbase + hi * 8;
    FragAB t;
    t.q[0] = *reinterpret_cast<const uint4*>(p);
    t.q[1] = *reinterpret_cast<const uint4*>(p + 16);
    return t.v;
}

// ---------------------------------------------------------------------------
// Workgroup GEMM: Out[64][N] = act(A[64][Kpad] * Wt^T + bias)
// A: bf16 row-major [64][Kpad];  Wt: bf16 [N rows][Kpad] (pre-transposed)
// act: 0=none, 1=sigmoid, 2=tanh.  8 waves split the 4*(N/16) tiles.
// ---------------------------------------------------------------------------
__device__ void mm_store(const bf16* A, int Kpad, const bf16* Wt, const float* bias,
                         float* Out, int N, int act) {
    int lane   = threadIdx.x & 31;
    int wave   = threadIdx.x >> 5;
    int nwaves = blockDim.x >> 5;
    int ntiles = N >> 4;
    int ktiles = Kpad >> 5;
    int total  = 4 * ntiles;   // M = 64 -> 4 row tiles
    for (int tidx = wave; tidx < total; tidx += nwaves) {
        int mt = tidx / ntiles;
        int nt = tidx - mt * ntiles;
        FragC acc;
        #pragma unroll
        for (int i = 0; i < 8; ++i) acc.f[i] = 0.f;
        const bf16* At = A  + (size_t)mt * 16 * Kpad;
        const bf16* Bt = Wt + (size_t)nt * 16 * Kpad;
        for (int kt = 0; kt < ktiles; ++kt) {
            v16bf af = load_frag(At, Kpad, lane, kt * 32);
            v16bf bf_ = load_frag(Bt, Kpad, lane, kt * 32);
            acc.v = __builtin_amdgcn_wmma_f32_16x16x32_bf16(
                false, af, false, bf_, (short)0, acc.v, false, false);
        }
        int col = nt * 16 + (lane & 15);
        int hi  = lane >> 4;
        float bv = bias ? bias[col] : 0.f;
        #pragma unroll
        for (int v = 0; v < 8; ++v) {
            int row = mt * 16 + v + hi * 8;          // C/D layout: VGPR v -> M = v (+8 hi half)
            float r = acc.f[v] + bv;
            if (act == 1)      r = 1.f / (1.f + __expf(-r));
            else if (act == 2) r = tanhf(r);
            Out[(size_t)row * N + col] = r;
        }
    }
}

// ---------------------------------------------------------------------------
// One GRU layer (TF GRUCell semantics) for the whole batch (64 rows).
// A already holds the x-part; this writes h / r*h into cols [hoff, hoff+128).
// ---------------------------------------------------------------------------
__device__ void gru_layer(bf16* A, int Kpad, int hoff,
                          const bf16* WgT, const float* gb,
                          const bf16* WcT, const float* cb,
                          float* G, float* C, float* h,
                          const int* seqlen, int t) {
    int tid = threadIdx.x, nth = blockDim.x;
    for (int i = tid; i < BB * HH; i += nth) {
        int b = i >> 7, c = i & 127;
        A[(size_t)b * Kpad + hoff + c] = (bf16)h[i];
    }
    __syncthreads();
    mm_store(A, Kpad, WgT, gb, G, 256, 1);             // gates: sigmoid -> r,u
    __syncthreads();
    for (int i = tid; i < BB * HH; i += nth) {
        int b = i >> 7, c = i & 127;
        A[(size_t)b * Kpad + hoff + c] = (bf16)(G[b * 256 + c] * h[i]);   // r*h
    }
    __syncthreads();
    mm_store(A, Kpad, WcT, cb, C, 128, 2);             // candidate: tanh
    __syncthreads();
    for (int i = tid; i < BB * HH; i += nth) {
        int b = i >> 7, c = i & 127;
        float u  = G[b * 256 + 128 + c];
        float nh = u * h[i] + (1.f - u) * C[i];
        if (seqlen && t >= seqlen[b]) nh = h[i];       // encoder copy-through
        h[i] = nh;
    }
    __syncthreads();
}

// ---------------------------------------------------------------------------
// Prep: fp32 [K][N] -> bf16 transposed [Npad][Kpad] (zero padded)
// ---------------------------------------------------------------------------
__global__ __launch_bounds__(256) void transpose_bf16_kernel(
        const float* __restrict__ src, bf16* __restrict__ dst,
        int K, int N, int Kpad, int Npad) {
    int idx = blockIdx.x * blockDim.x + threadIdx.x;
    int total = Kpad * Npad;
    if (idx >= total) return;
    int n = idx / Kpad, k = idx - n * Kpad;
    float v = (n < N && k < K) ? src[(size_t)k * N + n] : 0.f;
    dst[(size_t)n * Kpad + k] = (bf16)v;
}

// ---------------------------------------------------------------------------
// Embedding gather: exb/eyb layout [S][B][E] bf16 (per-step contiguous)
// ---------------------------------------------------------------------------
__global__ __launch_bounds__(256) void embed_kernel(
        const int* __restrict__ x, const int* __restrict__ y,
        const float* __restrict__ emb,
        bf16* __restrict__ exb, bf16* __restrict__ eyb) {
    int idx = blockIdx.x * blockDim.x + threadIdx.x;
    const int per = SS * BB * EE;
    if (idx >= 2 * per) return;
    int which = idx / per;
    int r = idx - which * per;
    int t = r / (BB * EE);
    int r2 = r - t * (BB * EE);
    int b = r2 / EE, e = r2 - b * EE;
    int tok = (which ? y : x)[b * SS + t];
    float v = emb[(size_t)tok * EE + e];
    (which ? eyb : exb)[r] = (bf16)v;
}

// ---------------------------------------------------------------------------
// Encoder: single workgroup, 2-layer GRU over 20 steps (WMMA per step)
// ---------------------------------------------------------------------------
__global__ __launch_bounds__(256) void encoder_kernel(
        const bf16* __restrict__ exb, const int* __restrict__ xlen,
        const bf16* e0gT, const float* e0gb, const bf16* e0cT, const float* e0cb,
        const bf16* e1gT, const float* e1gb, const bf16* e1cT, const float* e1cb,
        bf16* A, float* G, float* C, float* h0, float* h1,
        float* memF, bf16* memB, float* h0f, float* h1f) {
    int tid = threadIdx.x, nth = blockDim.x;
    for (int i = tid; i < BB * HH; i += nth) { h0[i] = 0.f; h1[i] = 0.f; }
    __syncthreads();
    for (int t = 0; t < SS; ++t) {
        // A = [ex_t | h0] padded to 448 (14 K-tiles)
        for (int i = tid; i < BB * EE; i += nth) {
            int b = i / EE, e = i - b * EE;
            A[(size_t)b * 448 + e] = exb[((size_t)t * BB + b) * EE + e];
        }
        for (int i = tid; i < BB * 20; i += nth) {
            int b = i / 20, c = i - b * 20;
            A[(size_t)b * 448 + 428 + c] = (bf16)0.f;
        }
        __syncthreads();
        gru_layer(A, 448, 300, e0gT, e0gb, e0cT, e0cb, G, C, h0, xlen, t);
        // layer 1: A = [n0 | h1], K = 256
        for (int i = tid; i < BB * HH; i += nth) {
            int b = i >> 7, c = i & 127;
            A[(size_t)b * 256 + c] = (bf16)h0[i];
        }
        __syncthreads();
        gru_layer(A, 256, 128, e1gT, e1gb, e1cT, e1cb, G, C, h1, xlen, t);
        // memory output (zeroed past seq end)
        for (int i = tid; i < BB * HH; i += nth) {
            int b = i >> 7, c = i & 127;
            float v = (t < xlen[b]) ? h1[i] : 0.f;
            memF[((size_t)b * SS + t) * HH + c] = v;
            memB[((size_t)b * SS + t) * HH + c] = (bf16)v;
        }
        __syncthreads();
    }
    for (int i = tid; i < BB * HH; i += nth) { h0f[i] = h0[i]; h1f[i] = h1[i]; }
}

// ---------------------------------------------------------------------------
// keys = memory @ W_mem : [1280,128]x[128,128], one wave per 16x16 tile
// ---------------------------------------------------------------------------
__global__ __launch_bounds__(256) void keys_kernel(
        const bf16* __restrict__ memB, const bf16* __restrict__ WmemT,
        float* __restrict__ keysF) {
    int lane = threadIdx.x & 31;
    int nt   = threadIdx.x >> 5;     // 8 N-tiles
    int mt   = blockIdx.x;           // 80 M-tiles
    FragC acc;
    #pragma unroll
    for (int i = 0; i < 8; ++i) acc.f[i] = 0.f;
    const bf16* At = memB  + (size_t)mt * 16 * HH;
    const bf16* Bt = WmemT + (size_t)nt * 16 * HH;
    #pragma unroll
    for (int kt = 0; kt < 4; ++kt) {
        v16bf af = load_frag(At, HH, lane, kt * 32);
        v16bf bf_ = load_frag(Bt, HH, lane, kt * 32);
        acc.v = __builtin_amdgcn_wmma_f32_16x16x32_bf16(
            false, af, false, bf_, (short)0, acc.v, false, false);
    }
    int col = nt * 16 + (lane & 15);
    int hi  = lane >> 4;
    #pragma unroll
    for (int v = 0; v < 8; ++v) {
        int row = mt * 16 + v + hi * 8;
        keysF[(size_t)row * HH + col] = acc.f[v];
    }
}

// ---------------------------------------------------------------------------
// Decoder: single workgroup, GRU + Bahdanau attention, emits na[t] (bf16)
// ---------------------------------------------------------------------------
__global__ __launch_bounds__(256) void decoder_kernel(
        const bf16* __restrict__ eyb, const int* __restrict__ xlen,
        const bf16* d0gT, const float* d0gb, const bf16* d0cT, const float* d0cb,
        const bf16* d1gT, const float* d1gb, const bf16* d1cT, const float* d1cb,
        const bf16* WqT, const bf16* WattnT, const float* __restrict__ v_att,
        const float* __restrict__ keysF, const float* __restrict__ memF,
        const float* __restrict__ h0f, const float* __restrict__ h1f,
        bf16* A, float* G, float* C, float* h0, float* h1, float* attn, float* SC,
        bf16* __restrict__ naT) {
    int tid = threadIdx.x, nth = blockDim.x;
    for (int i = tid; i < BB * HH; i += nth) { h0[i] = h0f[i]; h1[i] = h1f[i]; attn[i] = 0.f; }
    __syncthreads();
    for (int t = 0; t < SS; ++t) {
        // A = [ey_t | attn | h0] padded to 576 (18 K-tiles)
        for (int i = tid; i < BB * EE; i += nth) {
            int b = i / EE, e = i - b * EE;
            A[(size_t)b * 576 + e] = eyb[((size_t)t * BB + b) * EE + e];
        }
        for (int i = tid; i < BB * HH; i += nth) {
            int b = i >> 7, c = i & 127;
            A[(size_t)b * 576 + 300 + c] = (bf16)attn[i];
        }
        for (int i = tid; i < BB * 20; i += nth) {
            int b = i / 20, c = i - b * 20;
            A[(size_t)b * 576 + 556 + c] = (bf16)0.f;
        }
        __syncthreads();
        gru_layer(A, 576, 428, d0gT, d0gb, d0cT, d0cb, G, C, h0, nullptr, t);
        for (int i = tid; i < BB * HH; i += nth) {
            int b = i >> 7, c = i & 127;
            A[(size_t)b * 256 + c] = (bf16)h0[i];
        }
        __syncthreads();
        gru_layer(A, 256, 128, d1gT, d1gb, d1cT, d1cb, G, C, h1, nullptr, t);
        // q = n1 @ W_q  -> reuse G as Q[b*128+u]
        for (int i = tid; i < BB * HH; i += nth) {
            int b = i >> 7, c = i & 127;
            A[(size_t)b * 128 + c] = (bf16)h1[i];
        }
        __syncthreads();
        mm_store(A, 128, WqT, nullptr, G, 128, 0);
        __syncthreads();
        // scores: sc[b,s] = sum_u tanh(keys + q) * v_att, masked by x_length
        for (int p = tid; p < BB * SS; p += nth) {
            int b = p / SS, s = p - b * SS;
            const float* kr = keysF + ((size_t)b * SS + s) * HH;
            const float* qr = G + (size_t)b * HH;
            float acc = 0.f;
            for (int u = 0; u < HH; ++u) acc += tanhf(kr[u] + qr[u]) * v_att[u];
            SC[p] = (s < xlen[b]) ? acc : -3.402823466e38f;
        }
        __syncthreads();
        // softmax over S per batch row
        if (tid < BB) {
            float m = -3.402823466e38f;
            for (int s = 0; s < SS; ++s) m = fmaxf(m, SC[tid * SS + s]);
            float e[SS]; float sum = 0.f;
            for (int s = 0; s < SS; ++s) { e[s] = __expf(SC[tid * SS + s] - m); sum += e[s]; }
            float inv = 1.f / sum;
            for (int s = 0; s < SS; ++s) SC[tid * SS + s] = e[s] * inv;
        }
        __syncthreads();
        // ctx = al @ memory -> C
        for (int i = tid; i < BB * HH; i += nth) {
            int b = i >> 7, c = i & 127;
            float acc = 0.f;
            for (int s = 0; s < SS; ++s)
                acc += SC[b * SS + s] * memF[((size_t)b * SS + s) * HH + c];
            C[i] = acc;
        }
        __syncthreads();
        // na = [n1 | ctx] @ W_attn -> attn (carried), store bf16 row t*64+b
        for (int i = tid; i < BB * HH; i += nth) {
            int b = i >> 7, c = i & 127;
            A[(size_t)b * 256 + c]       = (bf16)h1[i];
            A[(size_t)b * 256 + 128 + c] = (bf16)C[i];
        }
        __syncthreads();
        mm_store(A, 256, WattnT, nullptr, attn, 128, 0);
        __syncthreads();
        for (int i = tid; i < BB * HH; i += nth) {
            int b = i >> 7, c = i & 127;
            naT[((size_t)t * BB + b) * HH + c] = (bf16)attn[i];
        }
        __syncthreads();
    }
}

// ---------------------------------------------------------------------------
// Projection: logits[b,t,:] = na[t,b] @ W_proj + b_proj (masked by y_length)
// [1280 x 50257], K=128. grid = (3142 N-tiles, 10), 8 waves * 16x16 each.
// ---------------------------------------------------------------------------
__global__ __launch_bounds__(256) void proj_kernel(
        const bf16* __restrict__ naT, const bf16* __restrict__ WprojT,
        const float* __restrict__ bproj, const int* __restrict__ ylen,
        float* __restrict__ out) {
    int lane = threadIdx.x & 31;
    int mt   = blockIdx.y * 8 + (threadIdx.x >> 5);   // 0..79
    int nt   = blockIdx.x;                             // 0..3141
    FragC acc;
    #pragma unroll
    for (int i = 0; i < 8; ++i) acc.f[i] = 0.f;
    const bf16* At = naT    + (size_t)mt * 16 * HH;
    const bf16* Bt = WprojT + (size_t)nt * 16 * HH;
    #pragma unroll
    for (int kt = 0; kt < 4; ++kt) {
        v16bf af = load_frag(At, HH, lane, kt * 32);
        v16bf bf_ = load_frag(Bt, HH, lane, kt * 32);
        acc.v = __builtin_amdgcn_wmma_f32_16x16x32_bf16(
            false, af, false, bf_, (short)0, acc.v, false, false);
    }
    int vcol = nt * 16 + (lane & 15);
    if (vcol >= VV) return;                            // edge tile (V % 16 == 1)
    int hi = lane >> 4;
    float bias = bproj[vcol];
    #pragma unroll
    for (int v = 0; v < 8; ++v) {
        int m = mt * 16 + v + hi * 8;                  // m = t*64 + b
        int t = m >> 6, b = m & 63;
        float val = (t < ylen[b]) ? (acc.f[v] + bias) : 0.f;
        out[((size_t)b * SS + t) * VV + vcol] = val;
    }
}

// ---------------------------------------------------------------------------
// Host launcher
// ---------------------------------------------------------------------------
extern "C" void kernel_launch(void* const* d_in, const int* in_sizes, int n_in,
                              void* d_out, int out_size, void* d_ws, size_t ws_size,
                              hipStream_t stream) {
    (void)in_sizes; (void)n_in; (void)out_size; (void)ws_size;

    const int*   x     = (const int*)  d_in[0];
    const int*   xlen  = (const int*)  d_in[1];
    const int*   y     = (const int*)  d_in[2];
    const int*   ylen  = (const int*)  d_in[3];
    const float* emb   = (const float*)d_in[4];
    const float* e0gk  = (const float*)d_in[5];  const float* e0gb = (const float*)d_in[6];
    const float* e0ck  = (const float*)d_in[7];  const float* e0cb = (const float*)d_in[8];
    const float* e1gk  = (const float*)d_in[9];  const float* e1gb = (const float*)d_in[10];
    const float* e1ck  = (const float*)d_in[11]; const float* e1cb = (const float*)d_in[12];
    const float* d0gk  = (const float*)d_in[13]; const float* d0gb = (const float*)d_in[14];
    const float* d0ck  = (const float*)d_in[15]; const float* d0cb = (const float*)d_in[16];
    const float* d1gk  = (const float*)d_in[17]; const float* d1gb = (const float*)d_in[18];
    const float* d1ck  = (const float*)d_in[19]; const float* d1cb = (const float*)d_in[20];
    const float* Wmem  = (const float*)d_in[21];
    const float* Wq    = (const float*)d_in[22];
    const float* vatt  = (const float*)d_in[23];
    const float* Wattn = (const float*)d_in[24];
    const float* Wproj = (const float*)d_in[25];
    const float* bproj = (const float*)d_in[26];
    float* out = (float*)d_out;

    // bump allocator inside d_ws (256B aligned)
    char* ws = (char*)d_ws;
    size_t off = 0;
    auto alloc = [&](size_t bytes) -> char* {
        size_t o = (off + 255) & ~(size_t)255;
        off = o + bytes;
        return ws + o;
    };

    bf16* wtE0g = (bf16*)alloc((size_t)256 * 448 * 2);
    bf16* wtE0c = (bf16*)alloc((size_t)128 * 448 * 2);
    bf16* wtE1g = (bf16*)alloc((size_t)256 * 256 * 2);
    bf16* wtE1c = (bf16*)alloc((size_t)128 * 256 * 2);
    bf16* wtD0g = (bf16*)alloc((size_t)256 * 576 * 2);
    bf16* wtD0c = (bf16*)alloc((size_t)128 * 576 * 2);
    bf16* wtD1g = (bf16*)alloc((size_t)256 * 256 * 2);
    bf16* wtD1c = (bf16*)alloc((size_t)128 * 256 * 2);
    bf16* wtMem = (bf16*)alloc((size_t)128 * 128 * 2);
    bf16* wtQ   = (bf16*)alloc((size_t)128 * 128 * 2);
    bf16* wtAtt = (bf16*)alloc((size_t)128 * 256 * 2);
    bf16* wtPrj = (bf16*)alloc((size_t)VROWS_PAD * 128 * 2);
    bf16* exb   = (bf16*)alloc((size_t)SS * BB * EE * 2);
    bf16* eyb   = (bf16*)alloc((size_t)SS * BB * EE * 2);
    float* memF = (float*)alloc((size_t)BB * SS * HH * 4);
    bf16*  memB = (bf16*)alloc((size_t)BB * SS * HH * 2);
    float* keysF= (float*)alloc((size_t)BB * SS * HH * 4);
    float* h0f  = (float*)alloc((size_t)BB * HH * 4);
    float* h1f  = (float*)alloc((size_t)BB * HH * 4);
    bf16*  naT  = (bf16*)alloc((size_t)SS * BB * HH * 2);
    // recurrence scratch (L2-resident activations)
    bf16*  A_s  = (bf16*)alloc((size_t)BB * 576 * 2);
    float* G_s  = (float*)alloc((size_t)BB * 256 * 4);
    float* C_s  = (float*)alloc((size_t)BB * HH * 4);
    float* h0_s = (float*)alloc((size_t)BB * HH * 4);
    float* h1_s = (float*)alloc((size_t)BB * HH * 4);
    float* at_s = (float*)alloc((size_t)BB * HH * 4);
    float* SC_s = (float*)alloc((size_t)BB * SS * 4);

    // 1) weight convert + transpose to bf16 [Npad][Kpad]
    auto T = [&](const float* src, bf16* dst, int K, int N, int Kpad, int Npad) {
        int total = Kpad * Npad;
        transpose_bf16_kernel<<<(total + 255) / 256, 256, 0, stream>>>(src, dst, K, N, Kpad, Npad);
    };
    T(e0gk, wtE0g, 428, 256, 448, 256);
    T(e0ck, wtE0c, 428, 128, 448, 128);
    T(e1gk, wtE1g, 256, 256, 256, 256);
    T(e1ck, wtE1c, 256, 128, 256, 128);
    T(d0gk, wtD0g, 556, 256, 576, 256);
    T(d0ck, wtD0c, 556, 128, 576, 128);
    T(d1gk, wtD1g, 256, 256, 256, 256);
    T(d1ck, wtD1c, 256, 128, 256, 128);
    T(Wmem, wtMem, 128, 128, 128, 128);
    T(Wq,   wtQ,   128, 128, 128, 128);
    T(Wattn,wtAtt, 256, 128, 256, 128);
    T(Wproj,wtPrj, 128, VV,  128, VROWS_PAD);

    // 2) embedding gather (bf16, [S][B][E])
    {
        int total = 2 * SS * BB * EE;
        embed_kernel<<<(total + 255) / 256, 256, 0, stream>>>(x, y, emb, exb, eyb);
    }

    // 3) encoder recurrence (single WGP, WMMA per step)
    encoder_kernel<<<1, 256, 0, stream>>>(exb, xlen,
        wtE0g, e0gb, wtE0c, e0cb, wtE1g, e1gb, wtE1c, e1cb,
        A_s, G_s, C_s, h0_s, h1_s, memF, memB, h0f, h1f);

    // 4) keys = memory @ W_mem (parallel WMMA)
    keys_kernel<<<80, 256, 0, stream>>>(memB, wtMem, keysF);

    // 5) decoder recurrence with attention (single WGP, WMMA per step)
    decoder_kernel<<<1, 256, 0, stream>>>(eyb, xlen,
        wtD0g, d0gb, wtD0c, d0cb, wtD1g, d1gb, wtD1c, d1cb,
        wtQ, wtAtt, vatt, keysF, memF, h0f, h1f,
        A_s, G_s, C_s, h0_s, h1_s, at_s, SC_s, naT);

    // 6) vocab projection: [1280 x 50257] WMMA GEMM, bandwidth-bound output
    {
        dim3 grid((VV + 15) / 16, 10);
        proj_kernel<<<grid, 256, 0, stream>>>(naT, wtPrj, bproj, ylen, out);
    }
}
